// MultilayerGRU_13846974562801
// MI455X (gfx1250) — matrix operand order repaired and checked by reading, not compiled, not run on hardware
//
#include <hip/hip_runtime.h>
#include <hip/hip_bf16.h>
#include <stdint.h>

// Problem sizes (fixed by the reference)
constexpr int B_ = 32, S_ = 512, I_ = 256, H_ = 1024, O_ = 256, L_ = 3;
constexpr unsigned GRU_BLOCKS = 32;   // persistent kernel blocks (8 waves each = 256 waves)

typedef __attribute__((ext_vector_type(16))) __bf16 v16bf;
typedef __attribute__((ext_vector_type(8)))  float  v8f;

union BFrag { v16bf v; uint4 q[2]; };

struct Frags { uint4 a0, a1, b0, b1; };

__device__ __forceinline__ uint16_t f2bf(float f) {
    uint32_t u = __float_as_uint(f);
    uint32_t r = (u + 0x7FFFu + ((u >> 16) & 1u)) >> 16;
    return (uint16_t)r;
}

__device__ __forceinline__ void load_frags(Frags& f, const uint16_t* ap, const uint16_t* bp) {
    f.a0 = *(const uint4*)ap;        // A chunk: K = k .. k+7   (this lane)
    f.a1 = *(const uint4*)(ap + 16); // A chunk: K = k+16 .. k+23
    f.b0 = *(const uint4*)bp;        // B chunk: K = k .. k+7   (this lane)
    f.b1 = *(const uint4*)(bp + 16);
}

__device__ __forceinline__ v8f wmma_acc(const Frags& f, v8f acc) {
    BFrag a, b;
    a.q[0] = f.a0; a.q[1] = f.a1;
    b.q[0] = f.b0; b.q[1] = f.b1;
    return __builtin_amdgcn_wmma_f32_16x16x32_bf16(false, a.v, false, b.v,
                                                   (short)0, acc, false, false);
}

// Software-pipelined K-segment GEMM: 4 in-flight fragment buffers, prefetch
// distance 128 K-elements (12 outstanding b128 loads ahead of each wait).
// klen must be a multiple of 128 (here: 1024 or 256).
__device__ __forceinline__ v8f gemm_seg(v8f acc, const uint16_t* __restrict__ a,
                                        const uint16_t* __restrict__ b, int klen) {
    Frags f0, f1, f2, f3;
    load_frags(f0, a +  0, b +  0);
    load_frags(f1, a + 32, b + 32);
    load_frags(f2, a + 64, b + 64);
    load_frags(f3, a + 96, b + 96);
    int kb = 0;
    for (; kb + 128 < klen; kb += 128) {
        acc = wmma_acc(f0, acc); load_frags(f0, a + kb + 128, b + kb + 128);
        acc = wmma_acc(f1, acc); load_frags(f1, a + kb + 160, b + kb + 160);
        acc = wmma_acc(f2, acc); load_frags(f2, a + kb + 192, b + kb + 192);
        acc = wmma_acc(f3, acc); load_frags(f3, a + kb + 224, b + kb + 224);
    }
    acc = wmma_acc(f0, acc);
    acc = wmma_acc(f1, acc);
    acc = wmma_acc(f2, acc);
    acc = wmma_acc(f3, acc);
    return acc;
}

// Fast gate activations: v_exp_f32 + v_rcp_f32 (error << bf16 rounding noise)
__device__ __forceinline__ float fast_sigmoid(float x) {
    return __builtin_amdgcn_rcpf(1.0f + __expf(-x));
}
__device__ __forceinline__ float fast_tanh(float x) {
    float e = __expf(-2.0f * x);
    return 1.0f - 2.0f * e * __builtin_amdgcn_rcpf(1.0f + e);
}

// Monotonic grid barrier (counter reset each launch by init kernel).
__device__ __forceinline__ void grid_sync(unsigned* ctr, unsigned& epoch, unsigned nblocks) {
    __threadfence();
    __syncthreads();
    if (threadIdx.x == 0) {
        epoch += nblocks;
        __hip_atomic_fetch_add(ctr, 1u, __ATOMIC_RELEASE, __HIP_MEMORY_SCOPE_AGENT);
        while (__hip_atomic_load(ctr, __ATOMIC_ACQUIRE, __HIP_MEMORY_SCOPE_AGENT) < epoch) {
            __builtin_amdgcn_s_sleep(1);
        }
    }
    __syncthreads();
    __threadfence();
}

// ---------------------------------------------------------------------------
// Init: convert x and Wout to bf16, seed per-layer h buffers from h0, reset ctr
__global__ void __launch_bounds__(256) init_kernel(
    const float* __restrict__ x, const float* __restrict__ h0,
    const float* __restrict__ Wout,
    uint16_t* __restrict__ x_bf, float* __restrict__ hbufs,
    uint16_t* __restrict__ hbf, uint16_t* __restrict__ Wout_bf,
    unsigned* __restrict__ ctr) {
    size_t tid = (size_t)blockIdx.x * blockDim.x + threadIdx.x;
    size_t stride = (size_t)gridDim.x * blockDim.x;
    if (tid == 0) *ctr = 0u;
    for (size_t i = tid; i < (size_t)B_ * S_ * I_; i += stride) x_bf[i] = f2bf(x[i]);
    for (size_t i = tid; i < (size_t)L_ * B_ * H_; i += stride) {
        int l = (int)(i >> 15); int b = (int)((i >> 10) & 31); int j = (int)(i & 1023);
        float v = h0[(size_t)b * L_ * H_ + (size_t)l * H_ + j];
        hbufs[i] = v; hbf[i] = f2bf(v);
    }
    for (size_t i = tid; i < (size_t)O_ * H_; i += stride) Wout_bf[i] = f2bf(Wout[i]);
}

// ---------------------------------------------------------------------------
// Pack one layer's weights into bf16 [N, Kcat] (K = [hidden | input]) matrices.
// Wzr: rows 0..2047 (gates z,r).  Wg: rows 0..1023 (gate g).
__global__ void __launch_bounds__(256) pack_weights(
    const float* __restrict__ Wh_l,   // [3,H,H]
    const float* __restrict__ Wx_l,   // [3,H,Kx]
    uint16_t* __restrict__ Wzr, uint16_t* __restrict__ Wg, int Kx) {
    const int Kcat = H_ + Kx;
    const size_t total = (size_t)3 * H_ * Kcat;
    size_t stride = (size_t)gridDim.x * blockDim.x;
    for (size_t idx = (size_t)blockIdx.x * blockDim.x + threadIdx.x; idx < total; idx += stride) {
        size_t r = idx / Kcat; int k = (int)(idx % Kcat);
        int g = (int)(r >> 10); int j = (int)(r & 1023);
        float v = (k < H_) ? Wh_l[((size_t)g * H_ + j) * H_ + k]
                           : Wx_l[((size_t)g * H_ + j) * Kx + (k - H_)];
        uint16_t bv = f2bf(v);
        if (g < 2) Wzr[r * Kcat + k] = bv;
        else       Wg[(size_t)j * Kcat + k] = bv;
    }
}

// ---------------------------------------------------------------------------
// Persistent GRU recurrence: 3 layers x 512 steps, 2 WMMA phases per step.
__global__ void __launch_bounds__(256, 1) gru_persistent(
    const float* __restrict__ bh0,    // [3,H]
    const float* __restrict__ bh12,   // [2,3,H]
    const uint16_t* __restrict__ Wzr0, const uint16_t* __restrict__ Wg0,
    const uint16_t* __restrict__ Wzr1, const uint16_t* __restrict__ Wg1,
    const uint16_t* __restrict__ Wzr2, const uint16_t* __restrict__ Wg2,
    const uint16_t* __restrict__ x_bf,          // [B,S,I] bf16
    uint16_t* __restrict__ hsA, uint16_t* __restrict__ hsB,  // [S,B,H] bf16
    float* __restrict__ hbufs,                  // [L,B,H] f32 running h
    uint16_t* __restrict__ hbf,                 // [L,B,H] bf16 running h
    float* __restrict__ z_buf,                  // [B,H] f32
    uint16_t* __restrict__ rh_bf,               // [B,H] bf16 (r*h)
    unsigned* __restrict__ ctr) {
    const int lane = threadIdx.x & 31;
    const int wid  = blockIdx.x * 8 + (threadIdx.x >> 5);   // 0..255
    const int m_l  = lane & 15;            // A row / B col / C col within tile
    const int koff = (lane >> 4) * 8;      // A-fragment K sub-offset
    const int kbB  = (lane >> 4) * 8;      // B-fragment K sub-offset
    const int mb   = (lane >> 4) * 8;      // C-fragment row base
    unsigned epoch = 0;

    for (int l = 0; l < L_; ++l) {
        const uint16_t* Wzr = (l == 0) ? Wzr0 : ((l == 1) ? Wzr1 : Wzr2);
        const uint16_t* Wg  = (l == 0) ? Wg0  : ((l == 1) ? Wg1  : Wg2);
        const int Kx   = (l == 0) ? I_ : H_;
        const int Kcat = H_ + Kx;
        const uint16_t* xin = (l == 0) ? x_bf : ((l == 1) ? hsA : hsB);
        uint16_t* hs_out = (l == 1) ? hsB : hsA;
        float*    hf = hbufs + (size_t)l * B_ * H_;
        uint16_t* hb = hbf   + (size_t)l * B_ * H_;
        const float* bias = (l == 0) ? bh0 : (bh12 + (size_t)(l - 1) * 3 * H_);

        for (int s = 0; s < S_; ++s) {
            // ---------------- phase 1: z and r gates (256 tile-waves) ------
            {
                const int mt = wid >> 7;         // batch tile 0..1
                const int nt = wid & 127;        // output tile over 2048
                const int n0 = nt * 16;
                const int gate = n0 >> 10;       // 0 = z, 1 = r
                const int ng = n0 + m_l;         // this lane's weight row
                const int brow = mt * 16 + m_l;  // this lane's batch row
                const uint16_t* arow_h = hb + (size_t)brow * H_ + koff;
                const uint16_t* arow_x = ((l == 0)
                    ? x_bf + (size_t)brow * S_ * I_ + (size_t)s * I_
                    : xin  + (size_t)s * B_ * H_ + (size_t)brow * H_) + koff;
                const uint16_t* bline = Wzr + (size_t)ng * Kcat + kbB;
                v8f acc = {0.f, 0.f, 0.f, 0.f, 0.f, 0.f, 0.f, 0.f};
                acc = gemm_seg(acc, arow_h, bline, H_);        // hidden part
                acc = gemm_seg(acc, arow_x, bline + H_, Kx);   // input part
                const int j = ng & (H_ - 1);
                const float bv = bias[gate * H_ + j];
#pragma unroll
                for (int i = 0; i < 8; ++i) {
                    const int b = mt * 16 + mb + i;
                    const size_t o = (size_t)b * H_ + j;
                    float sg = fast_sigmoid(acc[i] + bv);
                    if (gate == 0) z_buf[o] = sg;
                    else           rh_bf[o] = f2bf(sg * hf[o]);
                }
            }
            grid_sync(ctr, epoch, GRU_BLOCKS);
            // ---------------- phase 2: candidate gate + state update -------
            if (wid < 128) {
                const int mt = wid >> 6;
                const int jt = wid & 63;
                const int jg = jt * 16 + m_l;
                const int brow = mt * 16 + m_l;
                const uint16_t* arow_h = rh_bf + (size_t)brow * H_ + koff;
                const uint16_t* arow_x = ((l == 0)
                    ? x_bf + (size_t)brow * S_ * I_ + (size_t)s * I_
                    : xin  + (size_t)s * B_ * H_ + (size_t)brow * H_) + koff;
                const uint16_t* bline = Wg + (size_t)jg * Kcat + kbB;
                v8f acc = {0.f, 0.f, 0.f, 0.f, 0.f, 0.f, 0.f, 0.f};
                acc = gemm_seg(acc, arow_h, bline, H_);
                acc = gemm_seg(acc, arow_x, bline + H_, Kx);
                const float bv = bias[2 * H_ + jg];
#pragma unroll
                for (int i = 0; i < 8; ++i) {
                    const int b = mt * 16 + mb + i;
                    const size_t o = (size_t)b * H_ + jg;
                    float g = fast_tanh(acc[i] + bv);
                    float z = z_buf[o];
                    float hn = z * hf[o] + (1.0f - z) * g;
                    hf[o] = hn;
                    uint16_t hnb = f2bf(hn);
                    hb[o] = hnb;
                    hs_out[(size_t)s * B_ * H_ + o] = hnb;
                }
            }
            grid_sync(ctr, epoch, GRU_BLOCKS);
        }
    }
}

// ---------------------------------------------------------------------------
// Output projection: out[b,s,:] = hs2[s,b,:] @ Wout^T + bout
__global__ void __launch_bounds__(256) out_gemm(
    const uint16_t* __restrict__ hs,       // [S,B,H] bf16 (layer 2 output)
    const uint16_t* __restrict__ Wout_bf,  // [O,H] bf16
    const float* __restrict__ bout, float* __restrict__ out) {
    const int lane = threadIdx.x & 31;
    const int wid  = blockIdx.x * 8 + (threadIdx.x >> 5);  // 0..16383
    const int mt = wid >> 4;       // 0..1023 (rows of S*B)
    const int nt = wid & 15;       // 0..15   (cols of O)
    const int m_l = lane & 15, koff = (lane >> 4) * 8;
    const int kbB = (lane >> 4) * 8, mb = (lane >> 4) * 8;
    const int rg = mt * 16 + m_l;
    const uint16_t* arow = hs + (size_t)rg * H_ + koff;
    const int n = nt * 16 + m_l;
    const uint16_t* bline = Wout_bf + (size_t)n * H_ + kbB;
    v8f acc = {0.f, 0.f, 0.f, 0.f, 0.f, 0.f, 0.f, 0.f};
    acc = gemm_seg(acc, arow, bline, H_);
    const float bo = bout[n];
#pragma unroll
    for (int i = 0; i < 8; ++i) {
        int r = mt * 16 + mb + i;
        int s = r >> 5, b = r & 31;
        out[(size_t)b * S_ * O_ + (size_t)s * O_ + n] = acc[i] + bo;
    }
}

// ---------------------------------------------------------------------------
// Final hidden states: d_out[b,l,:] = h_final(layer l)
__global__ void __launch_bounds__(256) write_hidden(
    const float* __restrict__ hbufs, float* __restrict__ outh) {
    int i = blockIdx.x * blockDim.x + threadIdx.x;
    if (i < L_ * B_ * H_) {
        int l = i >> 15; int b = (i >> 10) & 31; int j = i & 1023;
        outh[(size_t)b * L_ * H_ + (size_t)l * H_ + j] = hbufs[i];
    }
}

// ---------------------------------------------------------------------------
extern "C" void kernel_launch(void* const* d_in, const int* in_sizes, int n_in,
                              void* d_out, int out_size, void* d_ws, size_t ws_size,
                              hipStream_t stream) {
    const float* x    = (const float*)d_in[0];
    const float* h0   = (const float*)d_in[1];
    const float* Wx0  = (const float*)d_in[2];
    const float* Wh0  = (const float*)d_in[3];
    const float* bh0  = (const float*)d_in[4];
    const float* Wx12 = (const float*)d_in[5];
    const float* Wh12 = (const float*)d_in[6];
    const float* bh12 = (const float*)d_in[7];
    const float* Wout = (const float*)d_in[8];
    const float* bout = (const float*)d_in[9];
    float* out = (float*)d_out;
    float* out_hidden = out + (size_t)B_ * S_ * O_;

    char* base = (char*)d_ws;
    size_t off = 0;
    auto take = [&](size_t bytes) -> void* {
        off = (off + 255) & ~(size_t)255;
        void* r = base + off;
        off += bytes;
        return r;
    };
    unsigned* ctr     = (unsigned*)take(256);
    uint16_t* x_bf    = (uint16_t*)take((size_t)B_ * S_ * I_ * 2);
    uint16_t* Wzr0    = (uint16_t*)take((size_t)2048 * (H_ + I_) * 2);
    uint16_t* Wg0     = (uint16_t*)take((size_t)1024 * (H_ + I_) * 2);
    uint16_t* Wzr1    = (uint16_t*)take((size_t)2048 * (2 * H_) * 2);
    uint16_t* Wg1     = (uint16_t*)take((size_t)1024 * (2 * H_) * 2);
    uint16_t* Wzr2    = (uint16_t*)take((size_t)2048 * (2 * H_) * 2);
    uint16_t* Wg2     = (uint16_t*)take((size_t)1024 * (2 * H_) * 2);
    uint16_t* Wout_bf = (uint16_t*)take((size_t)O_ * H_ * 2);
    uint16_t* hsA     = (uint16_t*)take((size_t)S_ * B_ * H_ * 2);
    uint16_t* hsB     = (uint16_t*)take((size_t)S_ * B_ * H_ * 2);
    float*    hbufs   = (float*)take((size_t)L_ * B_ * H_ * 4);
    uint16_t* hbf     = (uint16_t*)take((size_t)L_ * B_ * H_ * 2);
    float*    z_buf   = (float*)take((size_t)B_ * H_ * 4);
    uint16_t* rh_bf   = (uint16_t*)take((size_t)B_ * H_ * 2);

    init_kernel<<<4096, 256, 0, stream>>>(x, h0, Wout, x_bf, hbufs, hbf, Wout_bf, ctr);
    pack_weights<<<4096, 256, 0, stream>>>(Wh0, Wx0, Wzr0, Wg0, I_);
    pack_weights<<<4096, 256, 0, stream>>>(Wh12, Wx12, Wzr1, Wg1, H_);
    pack_weights<<<4096, 256, 0, stream>>>(Wh12 + (size_t)3 * H_ * H_,
                                           Wx12 + (size_t)3 * H_ * H_, Wzr2, Wg2, H_);
    gru_persistent<<<GRU_BLOCKS, 256, 0, stream>>>(bh0, bh12, Wzr0, Wg0, Wzr1, Wg1,
                                                   Wzr2, Wg2, x_bf, hsA, hsB, hbufs,
                                                   hbf, z_buf, rh_bf, ctr);
    out_gemm<<<2048, 256, 0, stream>>>(hsA, Wout_bf, bout, out);
    write_hidden<<<(L_ * B_ * H_ + 255) / 256, 256, 0, stream>>>(hbufs, out_hidden);
}
